// TransFusion3_83614423318581
// MI455X (gfx1250) — compile-verified, matrix operand order checked by reading, not run on hardware
//
#include <hip/hip_runtime.h>
#include <hip/hip_fp16.h>

// ---------------------------------------------------------------------------
// MI455X (gfx1250). All GEMMs: v_wmma_f32_16x16x32_f16 (f16 in, f32 accum).
// Fragment layouts per CDNA5 ISA 7.12.2:
//   A (16x32 f16): lane L<16 -> row M=L, elems K={0..7,16..23};
//                  lane L>=16 -> row M=L-16, elems K={8..15,24..31}
//   B kept transposed in LDS as [N][K] so the same loader serves both.
//   C/D (16x16 f32): vgpr i -> M = i + (lane>=16 ? 8 : 0), N = lane%16.
// Data path: f16 operands everywhere (Q/K/V, LN out, GELU intermediate) to
// halve HBM traffic; f32 only on the residual / LayerNorm stream.
// ---------------------------------------------------------------------------

typedef __attribute__((ext_vector_type(16))) _Float16 v16h;
typedef __attribute__((ext_vector_type(8)))  _Float16 v8h;
typedef __attribute__((ext_vector_type(8)))  float    v8f;

#define WMMA_F16(a, b, c) \
  __builtin_amdgcn_wmma_f32_16x16x32_f16(false, (a), false, (b), (short)0, (c), false, false)

__device__ inline v16h frag_load(const _Float16* p_row0, int ldr, int koff) {
  const int lane = threadIdx.x & 31;
  const _Float16* p = p_row0 + (lane & 15) * ldr + koff + ((lane & 16) ? 8 : 0);
  v8h lo = *(const v8h*)(p);
  v8h hi = *(const v8h*)(p + 16);
  v16h r;
#pragma unroll
  for (int i = 0; i < 8; ++i) { r[i] = lo[i]; r[i + 8] = hi[i]; }
  return r;
}

__device__ inline v16h frag_from(v8h lo, v8h hi) {
  v16h r;
#pragma unroll
  for (int i = 0; i < 8; ++i) { r[i] = lo[i]; r[i + 8] = hi[i]; }
  return r;
}

__device__ inline float gelu_exact(float x) {
  return 0.5f * x * (1.0f + erff(x * 0.70710678118654752f));
}

// ---------------------------------------------------------------------------
// WMMA GEMM, block tile 64x128, K-step 32, 128 threads = 4 waves (2x2),
// wave tile 32x64 = 2x4 fragments -> 8 WMMA per 6 fragment loads per K-step.
// A-gather (division-free; batch<=2): off = m*rstr + k*kstr + (m>=bdiv? bext:0)
//   NCHW activations: rstr=1, kstr=HW, bdiv=HW, bext=(C-1)*HW
//   row-major:        rstr=K, kstr=1,  bdiv=huge, bext=0
// ---------------------------------------------------------------------------
template <typename AT, typename OT>
__global__ __launch_bounds__(128) void k_wmma_gemm(
    const AT* __restrict__ A, const float* __restrict__ B,
    const float* __restrict__ bias, OT* __restrict__ C,
    int M, int N, int K,
    long a_rstride, long a_kstride, long a_bdiv, long a_bext,
    int gelu_flag) {
  __shared__ __align__(16) _Float16 ldsA[64 * 40];    // [m][k]
  __shared__ __align__(16) _Float16 ldsB[128 * 40];   // [n][k] (transposed)
  const int tid  = threadIdx.x;
  const int wave = tid >> 5;
  const int lane = tid & 31;
  const int m0 = blockIdx.y * 64;
  const int n0 = blockIdx.x * 128;
  const int wm = (wave >> 1) * 32;   // 0 | 32
  const int wn = (wave & 1) * 64;    // 0 | 64

  v8f acc[2][4] = {};

  for (int k0 = 0; k0 < K; k0 += 32) {
    // prefetch next B K-tile into L2 (global_prefetch_b8)
    if (k0 + 32 < K) {
      long kp = k0 + 32 + (tid >> 5);
      long np = n0 + (tid & 31) * 4;
      if (kp < K && np < N) __builtin_prefetch(&B[kp * (long)N + np], 0, 1);
    }
    // A tile 64x32 -> f16 LDS (division-free gather)
    for (int idx = tid; idx < 64 * 32; idx += 128) {
      int r = idx >> 5, c = idx & 31;
      long m = m0 + r, k = k0 + c;
      float v = 0.f;
      if (m < M && k < K)
        v = (float)A[m * a_rstride + k * a_kstride + (m >= a_bdiv ? a_bext : 0)];
      ldsA[r * 40 + c] = (_Float16)v;
    }
    // B tile 32x128 -> transposed f16 LDS, float4 global loads
    for (int idx = tid; idx < 32 * 32; idx += 128) {
      int kk = idx >> 5, n4 = (idx & 31) * 4;
      long k = k0 + kk, n = n0 + n4;
      float4 val = make_float4(0.f, 0.f, 0.f, 0.f);
      if (k < K && n + 3 < N) val = *(const float4*)(B + k * (long)N + n);
      ldsB[(n4 + 0) * 40 + kk] = (_Float16)val.x;
      ldsB[(n4 + 1) * 40 + kk] = (_Float16)val.y;
      ldsB[(n4 + 2) * 40 + kk] = (_Float16)val.z;
      ldsB[(n4 + 3) * 40 + kk] = (_Float16)val.w;
    }
    __syncthreads();

    v16h af[2], bf[4];
#pragma unroll
    for (int i = 0; i < 2; ++i) af[i] = frag_load(ldsA + (wm + i * 16) * 40, 40, 0);
#pragma unroll
    for (int j = 0; j < 4; ++j) bf[j] = frag_load(ldsB + (wn + j * 16) * 40, 40, 0);
#pragma unroll
    for (int i = 0; i < 2; ++i)
#pragma unroll
      for (int j = 0; j < 4; ++j)
        acc[i][j] = WMMA_F16(af[i], bf[j], acc[i][j]);
    __syncthreads();
  }

#pragma unroll
  for (int mi = 0; mi < 2; ++mi)
#pragma unroll
    for (int nj = 0; nj < 4; ++nj)
#pragma unroll
      for (int i = 0; i < 8; ++i) {
        int row = m0 + wm + mi * 16 + i + ((lane & 16) ? 8 : 0);
        int col = n0 + wn + nj * 16 + (lane & 15);
        if (row < M && col < N) {
          float v = acc[mi][nj][i] + bias[col];
          if (gelu_flag) v = gelu_exact(v);
          C[(long)row * N + col] = (OT)v;
        }
      }
}

// ---------------------------------------------------------------------------
// Window cross-attention (faithful: scale=1, NO softmax), fused residual:
//   qf[b,h,w,c] = x2[b,c,h,w] + (Q K^T) V
// Q/K/V are f16. K tile staged with global_load_async_to_lds_b128 (ASYNCcnt).
// One block (128 thr, 4 waves) per (window, head); 16 q-rows/wave/iter.
// ---------------------------------------------------------------------------
__global__ __launch_bounds__(128) void k_win_attn(
    const _Float16* __restrict__ qbuf, const _Float16* __restrict__ kbuf,
    const _Float16* __restrict__ vbuf, const float* __restrict__ x2,
    float* __restrict__ qf) {
  __shared__ __align__(16) _Float16 ldsK[144 * 40];        // [krow][d]
  __shared__ __align__(16) _Float16 ldsVT[32 * 168];       // [d][krow], pad 144->160
  __shared__ __align__(16) _Float16 ldsAttn[4 * 16 * 168]; // per-wave staging

  const int tid  = threadIdx.x;
  const int lane = tid & 31;
  const int wave = tid >> 5;
  const int win  = blockIdx.x;
  const int head = blockIdx.y;
  const int b  = win >> 6;
  const int wy = (win >> 3) & 7;
  const int wx = win & 7;

  // K: async DMA global->LDS, 16B per lane (rows are raw f16, 4 chunks/row)
  for (int idx = tid; idx < 144 * 4; idx += 128) {
    int r = idx >> 2, c4 = idx & 3;
    int ty = r / 12, tx = r % 12;
    long m1 = ((long)b * 96 + wy * 12 + ty) * 96 + wx * 12 + tx;
    const _Float16* gp = kbuf + m1 * 256 + head * 32 + c4 * 8;
    unsigned ldst = (unsigned)(unsigned long long)(const void*)&ldsK[r * 40 + c4 * 8];
    asm volatile("global_load_async_to_lds_b128 %0, %1, off"
                 :: "v"(ldst), "v"(gp) : "memory");
  }
  // V: transposed stage (d-major), zero-pad K dim 144..159
  for (int idx = tid; idx < 32 * 160; idx += 128) {
    int d = idx / 160, kr = idx % 160;
    _Float16 v = (_Float16)0.f;
    if (kr < 144) {
      int ty = kr / 12, tx = kr % 12;
      long m1 = ((long)b * 96 + wy * 12 + ty) * 96 + wx * 12 + tx;
      v = vbuf[m1 * 256 + head * 32 + d];
    }
    ldsVT[d * 168 + kr] = v;
  }
  _Float16* attnW = ldsAttn + wave * 16 * 168;
  for (int e = lane; e < 16 * 16; e += 32) {      // zero attn pad cols 144..159
    int r = e >> 4, c = e & 15;
    attnW[r * 168 + 144 + c] = (_Float16)0.f;
  }
  asm volatile("s_wait_asynccnt 0x0" ::: "memory");
  __syncthreads();

  for (int it = 0; it < 9; ++it) {
    const int qbase = it * 64 + wave * 16;
    {
      // Q A-fragment directly from global f16 (two global_load_b128)
      int qr = qbase + (lane & 15);
      int ty = qr / 24, tx = qr % 24;
      long mq = ((long)b * 192 + wy * 24 + ty) * 192 + wx * 24 + tx;
      const _Float16* qp = qbuf + mq * 256 + head * 32 + ((lane & 16) ? 8 : 0);
      v16h af = frag_from(*(const v8h*)qp, *(const v8h*)(qp + 16));
      // attn(16x144) = Q(16x32) @ K^T, staged to LDS as f16
#pragma unroll
      for (int nt = 0; nt < 9; ++nt) {
        v16h bf = frag_load(ldsK + nt * 16 * 40, 40, 0);
        v8f z = {};
        v8f a = WMMA_F16(af, bf, z);
#pragma unroll
        for (int i = 0; i < 8; ++i) {
          int r = i + ((lane & 16) ? 8 : 0);
          int c = nt * 16 + (lane & 15);
          attnW[r * 168 + c] = (_Float16)a[i];
        }
      }
    }
    __syncthreads();
    // o(16x32) = attn(16x160pad) @ V(160x32)
#pragma unroll
    for (int nt = 0; nt < 2; ++nt) {
      v8f acc = {};
#pragma unroll
      for (int kk = 0; kk < 160; kk += 32) {
        v16h a2 = frag_load(attnW, 168, kk);
        v16h b2 = frag_load(ldsVT + nt * 16 * 168, 168, kk);
        acc = WMMA_F16(a2, b2, acc);
      }
#pragma unroll
      for (int i = 0; i < 8; ++i) {
        int r = qbase + i + ((lane & 16) ? 8 : 0);
        int ty = r / 24, tx = r % 24;
        int hh = wy * 24 + ty, ww = wx * 24 + tx;
        int c = head * 32 + nt * 16 + (lane & 15);
        long oidx = (((long)b * 192 + hh) * 192 + ww) * 256 + c;
        float xv = x2[((long)b * 256 + c) * 36864 + hh * 192 + ww];
        qf[oidx] = acc[i] + xv;
      }
    }
    __syncthreads();
  }
}

// ---------------------------------------------------------------------------
// Row LayerNorm over 256 channels; one block per row; OT = f32 or f16.
// ---------------------------------------------------------------------------
template <typename OT>
__global__ __launch_bounds__(256) void k_layernorm(
    const float* __restrict__ in, OT* __restrict__ out,
    const float* __restrict__ g, const float* __restrict__ bb) {
  __shared__ float red[256];
  const long row = blockIdx.x;
  const int  t   = threadIdx.x;
  float x = in[row * 256 + t];
  red[t] = x; __syncthreads();
  for (int s = 128; s > 0; s >>= 1) { if (t < s) red[t] += red[t + s]; __syncthreads(); }
  float mean = red[0] * (1.f / 256.f); __syncthreads();
  float d = x - mean;
  red[t] = d * d; __syncthreads();
  for (int s = 128; s > 0; s >>= 1) { if (t < s) red[t] += red[t + s]; __syncthreads(); }
  float var = red[0] * (1.f / 256.f);
  out[row * 256 + t] = (OT)(d * rsqrtf(var + 1e-5f) * g[t] + bb[t]);
}

// Final: out_nchw = LN(sc + h2) with NHWC->NCHW store.
__global__ __launch_bounds__(256) void k_final_ln(
    const float* __restrict__ sc, const float* __restrict__ h2,
    const float* __restrict__ g, const float* __restrict__ bb,
    float* __restrict__ out) {
  __shared__ float red[256];
  const long row = blockIdx.x;              // (b, h, w) over 2*192*192
  const int  t   = threadIdx.x;             // c
  float x = sc[row * 256 + t] + h2[row * 256 + t];
  red[t] = x; __syncthreads();
  for (int s = 128; s > 0; s >>= 1) { if (t < s) red[t] += red[t + s]; __syncthreads(); }
  float mean = red[0] * (1.f / 256.f); __syncthreads();
  float d = x - mean;
  red[t] = d * d; __syncthreads();
  for (int s = 128; s > 0; s >>= 1) { if (t < s) red[t] += red[t + s]; __syncthreads(); }
  float var = red[0] * (1.f / 256.f);
  float y = d * rsqrtf(var + 1e-5f) * g[t] + bb[t];
  long b = row / 36864, hw = row % 36864;
  out[(b * 256 + t) * 36864 + hw] = y;
}

// ---------------------------------------------------------------------------
// Bilinear align_corners=True interpolation on NHWC; optional accumulate.
// ---------------------------------------------------------------------------
__global__ void k_interp_ac(const float* __restrict__ in, float* __restrict__ out,
                            int B, int C, int Hi, int Wi, int Ho, int Wo,
                            int accumulate) {
  long total = (long)B * Ho * Wo * C;
  for (long i = blockIdx.x * (long)blockDim.x + threadIdx.x; i < total;
       i += (long)gridDim.x * blockDim.x) {
    int c = (int)(i % C); long tt = i / C;
    int xo = (int)(tt % Wo); tt /= Wo;
    int yo = (int)(tt % Ho); int b = (int)(tt / Ho);
    float fy = (Ho > 1) ? (float)yo * (float)(Hi - 1) / (float)(Ho - 1) : 0.f;
    float fx = (Wo > 1) ? (float)xo * (float)(Wi - 1) / (float)(Wo - 1) : 0.f;
    int y0 = (int)floorf(fy); int y1 = min(y0 + 1, Hi - 1); float wy = fy - (float)y0;
    int x0 = (int)floorf(fx); int x1 = min(x0 + 1, Wi - 1); float wx = fx - (float)x0;
    auto at = [&](int y, int x) { return in[(((long)b * Hi + y) * Wi + x) * C + c]; };
    float v = (at(y0, x0) * (1.f - wy) + at(y1, x0) * wy) * (1.f - wx) +
              (at(y0, x1) * (1.f - wy) + at(y1, x1) * wy) * wx;
    long o = (((long)b * Ho + yo) * Wo + xo) * C + c;
    if (accumulate) out[o] += v; else out[o] = v;
  }
}

// ---------------------------------------------------------------------------
// Avg-pool NCHW -> (B*64, C) tokens + faithful _sine_pos(F=8, Hp=8, Wp=C).
// ---------------------------------------------------------------------------
__device__ inline float sine_pos_val(int f, int yy, int xx, int Wp) {
  const float scale = 6.2831853071795864769f;
  int j; float e;
  if (f < 4) { j = f;     e = (float)(yy + 1) / (8.0f + 1e-5f) * scale; }
  else       { j = f - 4; e = (float)(xx + 1) / ((float)Wp + 1e-5f) * scale; }
  float dim_t = powf(10000.f, 2.f * (float)(j >> 1) / 4.f);
  float p = e / dim_t;
  return (j & 1) ? cosf(p) : sinf(p);
}

__global__ void k_pool_pos(const float* __restrict__ src, float* __restrict__ dst,
                           int B, int C, int H, int W, int wh, int ww) {
  long total = (long)B * 64 * C;
  for (long i = blockIdx.x * (long)blockDim.x + threadIdx.x; i < total;
       i += (long)gridDim.x * blockDim.x) {
    int c = (int)(i % C); long t = i / C;
    int x = (int)(t % 8); t /= 8;
    int y = (int)(t % 8); int b = (int)(t / 8);
    const float* sp = src + ((long)b * C + c) * H * W + (long)(y * wh) * W + x * ww;
    float s = 0.f;
    for (int dy = 0; dy < wh; ++dy)
      for (int dx = 0; dx < ww; ++dx) s += sp[dy * W + dx];
    s /= (float)(wh * ww);
    dst[i] = s + sine_pos_val(y, x, c, C);
  }
}

// ---------------------------------------------------------------------------
// Pooled global attention (64 tokens): softmax(QK^T) @ V per (b, head).
// ---------------------------------------------------------------------------
__global__ __launch_bounds__(64) void k_global_attn(
    const float* __restrict__ qg2, const float* __restrict__ kg2,
    const float* __restrict__ vg, float* __restrict__ og) {
  const int b = blockIdx.x, head = blockIdx.y;
  const int r = threadIdx.x;
  const float* qp = qg2 + ((long)b * 64 + r) * 256 + head * 32;
  float logits[64];
  float mx = -1e30f;
  for (int j = 0; j < 64; ++j) {
    const float* kp = kg2 + ((long)b * 64 + j) * 256 + head * 32;
    float d = 0.f;
    for (int t = 0; t < 32; ++t) d += qp[t] * kp[t];
    logits[j] = d; mx = fmaxf(mx, d);
  }
  float denom = 0.f;
  for (int j = 0; j < 64; ++j) { logits[j] = __expf(logits[j] - mx); denom += logits[j]; }
  float inv = 1.f / denom;
  for (int d = 0; d < 32; ++d) {
    float s = 0.f;
    for (int j = 0; j < 64; ++j)
      s += logits[j] * vg[((long)b * 64 + j) * 256 + head * 32 + d];
    og[((long)b * 64 + r) * 256 + head * 32 + d] = s * inv;
  }
}

// ---------------------------------------------------------------------------
extern "C" void kernel_launch(void* const* d_in, const int* in_sizes, int n_in,
                              void* d_out, int out_size, void* d_ws, size_t ws_size,
                              hipStream_t stream) {
  const float* x1     = (const float*)d_in[0];   // (2,512,96,96)
  const float* x2     = (const float*)d_in[1];   // (2,256,192,192)
  const float* q_w    = (const float*)d_in[2];
  const float* q_b    = (const float*)d_in[3];
  const float* k_w    = (const float*)d_in[4];
  const float* k_b    = (const float*)d_in[5];
  const float* v_w    = (const float*)d_in[6];
  const float* v_b    = (const float*)d_in[7];
  const float* q2_w   = (const float*)d_in[8];
  const float* q2_b   = (const float*)d_in[9];
  const float* k2_w   = (const float*)d_in[10];
  const float* k2_b   = (const float*)d_in[11];
  const float* v2_w   = (const float*)d_in[12];
  const float* v2_b   = (const float*)d_in[13];
  const float* proj_w = (const float*)d_in[14];
  const float* proj_b = (const float*)d_in[15];
  const float* proj_g = (const float*)d_in[16];
  const float* proj_be= (const float*)d_in[17];
  const float* n1_g   = (const float*)d_in[18];
  const float* n1_b   = (const float*)d_in[19];
  const float* n2_g   = (const float*)d_in[20];
  const float* n2_b   = (const float*)d_in[21];
  const float* fc1_w  = (const float*)d_in[22];
  const float* fc1_b  = (const float*)d_in[23];
  const float* fc2_w  = (const float*)d_in[24];
  const float* fc2_b  = (const float*)d_in[25];
  float* out = (float*)d_out;

  // workspace carve
  const long HW1 = 9216, HW2 = 36864;
  const long SCPRE_N = 2 * HW1 * 256;   // 4,718,592
  const long SC_N    = 2 * HW2 * 256;   // 18,874,368
  char* wsb = (char*)d_ws;
  auto carve_f = [&](long n) { float* p = (float*)wsb; wsb += n * sizeof(float); return p; };
  auto carve_h = [&](long n) { _Float16* p = (_Float16*)wsb; wsb += n * sizeof(_Float16); return p; };
  float*    scpre = carve_f(SCPRE_N);
  float*    sc    = carve_f(SC_N);
  float*    qf    = carve_f(SC_N);           // residual stream / fc2 out
  _Float16* qbuf  = carve_h(SC_N);           // 73728 x 256 f16
  _Float16* kbuf  = carve_h(SCPRE_N);        // 18432 x 256 f16
  _Float16* vbuf  = carve_h(SCPRE_N);
  _Float16* hbuf  = carve_h(SC_N);           // LN(qf) f16
  _Float16* h1    = carve_h(2L * HW2 * 512); // fc1 out f16
  float*    qg_s  = carve_f(128 * 256);
  float*    kg_s  = carve_f(128 * 512);
  float*    qg2   = carve_f(128 * 256);
  float*    kg2   = carve_f(128 * 256);
  float*    vg    = carve_f(128 * 256);
  float*    og    = carve_f(128 * 256);

  const long HUGE_BD = 1L << 60;
  auto grid2 = [](int M, int N) { return dim3((unsigned)((N + 127) / 128),
                                              (unsigned)((M + 63) / 64)); };

  // ---- skip branch: sc = interp(LN(x1n @ proj_w + proj_b)) ----
  k_wmma_gemm<float, float><<<grid2(18432, 256), 128, 0, stream>>>(
      x1, proj_w, proj_b, scpre, 18432, 256, 512, 1, HW1, HW1, 511 * HW1, 0);
  k_layernorm<float><<<18432, 256, 0, stream>>>(scpre, scpre, proj_g, proj_be);
  {
    long total = 2L * HW2 * 256;
    k_interp_ac<<<(int)((total + 255) / 256), 256, 0, stream>>>(
        scpre, sc, 2, 256, 96, 96, 192, 192, 0);
  }

  // ---- window attention: Q/K/V projections straight from NCHW, f16 out ----
  k_wmma_gemm<float, _Float16><<<grid2(73728, 256), 128, 0, stream>>>(
      x2, q_w, q_b, qbuf, 73728, 256, 256, 1, HW2, HW2, 255 * HW2, 0);
  k_wmma_gemm<float, _Float16><<<grid2(18432, 256), 128, 0, stream>>>(
      x1, k_w, k_b, kbuf, 18432, 256, 512, 1, HW1, HW1, 511 * HW1, 0);
  k_wmma_gemm<float, _Float16><<<grid2(18432, 256), 128, 0, stream>>>(
      x1, v_w, v_b, vbuf, 18432, 256, 512, 1, HW1, HW1, 511 * HW1, 0);
  {
    dim3 grid(128, 8);   // (window, head)
    k_win_attn<<<grid, 128, 0, stream>>>(qbuf, kbuf, vbuf, x2, qf);
  }

  // ---- pooled global branch ----
  {
    long tq = 2L * 64 * 256, tk = 2L * 64 * 512;
    k_pool_pos<<<(int)((tq + 255) / 256), 256, 0, stream>>>(x2, qg_s, 2, 256, 192, 192, 24, 24);
    k_pool_pos<<<(int)((tk + 255) / 256), 256, 0, stream>>>(x1, kg_s, 2, 512, 96, 96, 12, 12);
  }
  k_wmma_gemm<float, float><<<grid2(128, 256), 128, 0, stream>>>(
      qg_s, q2_w, q2_b, qg2, 128, 256, 256, 256, 1, HUGE_BD, 0, 0);
  k_wmma_gemm<float, float><<<grid2(128, 256), 128, 0, stream>>>(
      kg_s, k2_w, k2_b, kg2, 128, 256, 512, 512, 1, HUGE_BD, 0, 0);
  k_wmma_gemm<float, float><<<grid2(128, 256), 128, 0, stream>>>(
      kg_s, v2_w, v2_b, vg, 128, 256, 512, 512, 1, HUGE_BD, 0, 0);
  {
    dim3 grid(2, 8);
    k_global_attn<<<grid, 64, 0, stream>>>(qg2, kg2, vg, og);
  }
  {
    long total = 2L * HW2 * 256;   // qf += interp(og)
    k_interp_ac<<<(int)((total + 255) / 256), 256, 0, stream>>>(
        og, qf, 2, 256, 8, 8, 192, 192, 1);
  }

  // ---- MLP (f16 operands): h = GELU(LN(qf)@fc1)@fc2 ; out = LN(sc+h) ----
  k_layernorm<_Float16><<<73728, 256, 0, stream>>>(qf, hbuf, n1_g, n1_b);
  k_wmma_gemm<_Float16, _Float16><<<grid2(73728, 512), 128, 0, stream>>>(
      hbuf, fc1_w, fc1_b, h1, 73728, 512, 256, 256, 1, HUGE_BD, 0, 1);
  k_wmma_gemm<_Float16, float><<<grid2(73728, 256), 128, 0, stream>>>(
      h1, fc2_w, fc2_b, qf, 73728, 256, 512, 512, 1, HUGE_BD, 0, 0);
  k_final_ln<<<73728, 256, 0, stream>>>(sc, qf, n2_g, n2_b, out);
  (void)n_in; (void)in_sizes; (void)out_size; (void)ws_size;
}
